// ResonanceAttention_13872744366786
// MI455X (gfx1250) — matrix operand order repaired and checked by reading, not compiled, not run on hardware
//
#include <hip/hip_runtime.h>
#include <math.h>

// ---------------- problem constants ----------------
#define BATCH 4
#define SEQ   4096
#define DIM   256
#define XPITCH 264     // bf16 pitch for key tile in LDS (b128 frag reads conflict-minimal)
#define VPITCH 40      // bf16 pitch for VT tile / attn scratch in LDS
#define CTILE  32      // key/value columns per iteration (== bf16 WMMA K)
#define WAVES  4       // waves per block in attention kernel (64 query rows/block)

typedef __attribute__((ext_vector_type(2)))  float   v2f;
typedef __attribute__((ext_vector_type(8)))  float   v8f;
typedef __attribute__((ext_vector_type(8)))  __bf16  v8bf;
typedef __attribute__((ext_vector_type(16))) __bf16  v16bf;

__device__ __forceinline__ v8f zero8() {
    v8f z = {0.f,0.f,0.f,0.f,0.f,0.f,0.f,0.f};
    return z;
}
__device__ __forceinline__ v16bf cat16(v8bf lo, v8bf hi) {
    return __builtin_shufflevector(lo, hi, 0,1,2,3,4,5,6,7,8,9,10,11,12,13,14,15);
}
__device__ __forceinline__ v8bf cvt8(float4 a, float4 b) {
    v8bf r;
    r[0]=(__bf16)a.x; r[1]=(__bf16)a.y; r[2]=(__bf16)a.z; r[3]=(__bf16)a.w;
    r[4]=(__bf16)b.x; r[5]=(__bf16)b.y; r[6]=(__bf16)b.z; r[7]=(__bf16)b.w;
    return r;
}
__device__ __forceinline__ v8f wmma_f32(v2f a, v2f b, v8f c) {
    return __builtin_amdgcn_wmma_f32_16x16x4_f32(false, a, false, b, (short)0, c, false, false);
}
__device__ __forceinline__ v8f wmma_bf16(v16bf a, v16bf b, v8f c) {
    return __builtin_amdgcn_wmma_f32_16x16x32_bf16(false, a, false, b, (short)0, c, false, false);
}

// ---------------- kernel 1: sq[b,s] = sum_d x^2  (wave per row, fp32 exact) ----------------
__global__ void sq_rows_kernel(const float* __restrict__ x, float* __restrict__ sq) {
    int row  = blockIdx.x * (blockDim.x >> 5) + (threadIdx.x >> 5);
    int lane = threadIdx.x & 31;
    if (row >= BATCH * SEQ) return;
    const float* xr = x + (size_t)row * DIM;
    float acc = 0.f;
    #pragma unroll
    for (int j = 0; j < DIM / 32; ++j) {
        float v = xr[lane + 32 * j];
        acc = fmaf(v, v, acc);
    }
    acc += __shfl_xor(acc, 1, 32);
    acc += __shfl_xor(acc, 2, 32);
    acc += __shfl_xor(acc, 4, 32);
    acc += __shfl_xor(acc, 8, 32);
    acc += __shfl_xor(acc, 16, 32);
    if (lane == 0) sq[row] = acc;
}

// ---------------- kernel 2: VT[b][d][t] = bf16( x @ W^T + b ) ----------------
// fp32 WMMA GEMM (accurate pre-rounding), stored TRANSPOSED in bf16 so the
// attention kernel's B-operand K-pairs (along t) are memory-contiguous.
__global__ void v_gemm_kernel(const float* __restrict__ x,
                              const float* __restrict__ W,
                              const float* __restrict__ bias,
                              __bf16* __restrict__ VT) {
    int wid  = blockIdx.x * (blockDim.x >> 5) + (threadIdx.x >> 5);
    int lane = threadIdx.x & 31;
    int half = lane >> 4;
    int l16  = lane & 15;
    int ntile = wid & (DIM / 16 - 1);
    int mtile = wid >> 4;
    int m0 = mtile * 16, n0 = ntile * 16;

    const float* ap = x + (size_t)(m0 + l16) * DIM + half * 2;
    const float* bp = W + (size_t)(n0 + l16) * DIM + half * 2;
    v8f c = zero8();
    #pragma unroll 8
    for (int k0 = 0; k0 < DIM; k0 += 4) {
        v2f a, b;
        a.x = ap[k0]; a.y = ap[k0 + 1];
        b.x = bp[k0]; b.y = bp[k0 + 1];
        c = wmma_f32(a, b, c);
    }
    float bv = bias[n0 + l16];
    int mrow = m0 + half * 8;              // 8 consecutive t values per lane
    int bidx = mrow / SEQ;                 // tiles never straddle batches
    int tloc = mrow - bidx * SEQ;
    v8bf o;
    #pragma unroll
    for (int i = 0; i < 8; ++i) o[i] = (__bf16)(c[i] + bv);
    *(v8bf*)(VT + ((size_t)bidx * DIM + (n0 + l16)) * SEQ + tloc) = o;
}

// ---------------- kernel 3: fused Cauchy attention (bf16 WMMA, fp32 accumulate) ----------------
// 4 waves x 16 query rows; query A-fragments live in registers (64 VGPR/lane);
// key tile and V^T tile staged in LDS and shared by all 4 waves.
__launch_bounds__(WAVES * 32)
__global__ void cauchy_attn_kernel(const float* __restrict__ x,
                                   const __bf16* __restrict__ VT,
                                   const float* __restrict__ sq,
                                   const unsigned char* __restrict__ mask,
                                   const float* __restrict__ log_temp,
                                   float* __restrict__ out) {
    __shared__ __align__(16) __bf16 xcLds[CTILE * XPITCH];       // key tile (shared)
    __shared__ __align__(16) __bf16 vtLds[DIM * VPITCH];         // V^T tile (shared)
    __shared__ __align__(16) __bf16 atLds[WAVES][16 * VPITCH];   // attn C->A round trip

    const int tid  = threadIdx.x;
    const int wave = tid >> 5;
    const int lane = tid & 31;
    const int half = lane >> 4;
    const int l16  = lane & 15;
    const int b    = blockIdx.y;
    const int row0 = blockIdx.x * (WAVES * 16) + wave * 16;

    const float*  xb  = x  + (size_t)b * SEQ * DIM;
    const __bf16* vtb = VT + (size_t)b * DIM * SEQ;
    const float*  sqb = sq + (size_t)b * SEQ;
    const unsigned char* mb = mask + (size_t)b * SEQ;

    float t = expf(log_temp[0]);
    t = fmaxf(t, 1e-5f);
    const float invt2 = 1.0f / (t * t);

    // ---- load this wave's query strip straight into A-fragment registers ----
    // lane<16 : row row0+l16, K = k0+0..7  and k0+16..23
    // lane>=16: row row0+l16, K = k0+8..15 and k0+24..31   (half*8 base shift)
    v16bf xqA[DIM / 32];
    {
        const float* xrow = xb + (size_t)(row0 + l16) * DIM + half * 8;
        #pragma unroll
        for (int ks = 0; ks < DIM / 32; ++ks) {
            const float4* p0 = (const float4*)(xrow + ks * 32);
            const float4* p1 = (const float4*)(xrow + ks * 32 + 16);
            xqA[ks] = cat16(cvt8(p0[0], p0[1]), cvt8(p1[0], p1[1]));
        }
    }

    float sqr[8];
    #pragma unroll
    for (int i = 0; i < 8; ++i) sqr[i] = sqb[row0 + half * 8 + i];

    v8f oacc[DIM / 16];
    #pragma unroll
    for (int n = 0; n < DIM / 16; ++n) oacc[n] = zero8();
    float lsum[8] = {0.f,0.f,0.f,0.f,0.f,0.f,0.f,0.f};

    __bf16* atp = atLds[wave];

    for (int ct = 0; ct < SEQ / CTILE; ++ct) {
        const int col0 = ct * CTILE;
        __syncthreads();   // previous iteration done with xcLds / vtLds
        // stage key tile (32 rows x 256 f32 -> bf16); 128 threads
        for (int idx = tid; idx < CTILE * (DIM / 8); idx += WAVES * 32) {
            int r = idx >> 5, c8 = idx & 31;
            const float4* src = (const float4*)(xb + (size_t)(col0 + r) * DIM + c8 * 8);
            *(v8bf*)&xcLds[r * XPITCH + c8 * 8] = cvt8(src[0], src[1]);
        }
        // stage V^T tile (256 d-rows x 32 t) -- already bf16
        for (int idx = tid; idx < DIM * (CTILE / 8); idx += WAVES * 32) {
            int d = idx >> 2, p = idx & 3;
            *(v8bf*)&vtLds[d * VPITCH + p * 8] =
                *(const v8bf*)(vtb + (size_t)d * SEQ + col0 + p * 8);
        }
        // prefetch next tile into cache hierarchy (global_prefetch_b8)
        if (ct + 1 < SEQ / CTILE) {
            __builtin_prefetch(xb + (size_t)(col0 + CTILE) * DIM + (size_t)tid * 64, 0, 0);
            __builtin_prefetch(vtb + (size_t)(tid * 2) * SEQ + col0 + CTILE, 0, 0);
        }
        __syncthreads();

        // ---- Gram: two 16x16 tiles, K-loop of 8 bf16 WMMAs each; A from registers ----
        v8f g0 = zero8(), g1 = zero8();
        {
            const __bf16* b0 = &xcLds[l16 * XPITCH + half * 8];
            const __bf16* b1 = &xcLds[(16 + l16) * XPITCH + half * 8];
            #pragma unroll
            for (int ks = 0; ks < DIM / 32; ++ks) {
                int k0 = ks * 32;
                v16bf bb0 = cat16(*(const v8bf*)(b0 + k0), *(const v8bf*)(b0 + k0 + 16));
                v16bf bb1 = cat16(*(const v8bf*)(b1 + k0), *(const v8bf*)(b1 + k0 + 16));
                g0 = wmma_bf16(xqA[ks], bb0, g0);
                g1 = wmma_bf16(xqA[ks], bb1, g1);
            }
        }

        // ---- elementwise Cauchy kernel; exact-zero diagonal; stash bf16 attn tile ----
        // (row mask is applied once in the epilogue -- identical result)
        const float sqc0 = sqb[col0 + l16];
        const float sqc1 = sqb[col0 + 16 + l16];
        const float mc0  = mb[col0 + l16] ? 1.0f : 0.0f;
        const float mc1  = mb[col0 + 16 + l16] ? 1.0f : 0.0f;
        #pragma unroll
        for (int i = 0; i < 8; ++i) {
            const int rg = row0 + half * 8 + i;
            float d20 = fmaxf(sqr[i] + sqc0 - 2.0f * g0[i], 0.0f);
            float d21 = fmaxf(sqr[i] + sqc1 - 2.0f * g1[i], 0.0f);
            if (rg == col0 + l16)      d20 = 0.0f;   // exact diagonal (cancellation-safe)
            if (rg == col0 + 16 + l16) d21 = 0.0f;
            // result feeds a bf16 round -> hardware v_rcp_f32 is plenty accurate
            float kv0 = mc0 * __builtin_amdgcn_rcpf(fmaf(d20, invt2, 1.0f));
            float kv1 = mc1 * __builtin_amdgcn_rcpf(fmaf(d21, invt2, 1.0f));
            lsum[i] += kv0 + kv1;
            atp[(half * 8 + i) * VPITCH + l16]      = (__bf16)kv0;  // C-layout store
            atp[(half * 8 + i) * VPITCH + 16 + l16] = (__bf16)kv1;
        }

        // ---- attn(16x32) @ V(32x256): one A-frag, 16 N-tiles of bf16 WMMA ----
        {
            v16bf aat = cat16(*(const v8bf*)&atp[l16 * VPITCH + half * 8],
                              *(const v8bf*)&atp[l16 * VPITCH + 16 + half * 8]);
            #pragma unroll
            for (int n = 0; n < DIM / 16; ++n) {
                const __bf16* vp = &vtLds[(n * 16 + l16) * VPITCH + half * 8];
                v16bf bb = cat16(*(const v8bf*)vp, *(const v8bf*)(vp + 16));
                oacc[n] = wmma_bf16(aat, bb, oacc[n]);
            }
        }
    }

    // ---- row-sum reduction (16-lane halves) + normalize + row mask + store ----
    #pragma unroll
    for (int i = 0; i < 8; ++i) {
        float v = lsum[i];
        v += __shfl_xor(v, 1, 32);
        v += __shfl_xor(v, 2, 32);
        v += __shfl_xor(v, 4, 32);
        v += __shfl_xor(v, 8, 32);
        float mr = mb[row0 + half * 8 + i] ? 1.0f : 0.0f;
        lsum[i] = mr * __builtin_amdgcn_rcpf(fmaxf(v, 1e-8f));
    }
    float* ob = out + ((size_t)b * SEQ + row0) * DIM;
    #pragma unroll
    for (int n = 0; n < DIM / 16; ++n) {
        #pragma unroll
        for (int i = 0; i < 8; ++i) {
            ob[(size_t)(half * 8 + i) * DIM + n * 16 + l16] = oacc[n][i] * lsum[i];
        }
    }
}

// ---------------- host launcher ----------------
extern "C" void kernel_launch(void* const* d_in, const int* in_sizes, int n_in,
                              void* d_out, int out_size, void* d_ws, size_t ws_size,
                              hipStream_t stream) {
    const float*         x    = (const float*)d_in[0];
    const float*         W_v  = (const float*)d_in[1];
    const float*         b_v  = (const float*)d_in[2];
    const float*         lt   = (const float*)d_in[3];
    const unsigned char* mask = (const unsigned char*)d_in[4];
    float* out = (float*)d_out;

    // workspace: VT bf16 [B*D*S] (8 MB), then sq fp32 [B*S] (64 KB)
    __bf16* VT = (__bf16*)d_ws;
    float*  sq = (float*)(VT + (size_t)BATCH * DIM * SEQ);

    sq_rows_kernel<<<(BATCH * SEQ) / 8, 256, 0, stream>>>(x, sq);

    {
        int tiles = (BATCH * SEQ / 16) * (DIM / 16);
        v_gemm_kernel<<<tiles / 2, 64, 0, stream>>>(x, W_v, b_v, VT);
    }

    {
        dim3 grid(SEQ / (WAVES * 16), BATCH);
        cauchy_attn_kernel<<<grid, WAVES * 32, 0, stream>>>(x, VT, sq, mask, lt, out);
    }
}
